// EncoderCRF_47313359732815
// MI455X (gfx1250) — matrix operand order repaired and checked by reading, not compiled, not run on hardware
//
#include <hip/hip_runtime.h>
#include <hip/hip_bf16.h>

typedef __bf16 bf16_t;
typedef __attribute__((ext_vector_type(16))) __bf16 v16bf;
typedef __attribute__((ext_vector_type(8)))  __bf16 v8bf;
typedef __attribute__((ext_vector_type(8)))  float  v8f;

#define S_LEN 16384
#define E_DIM 512
#define HD    512
#define H_DIM 1024
#define G3    1536          // 3*HD
#define NTAG  64
#define START_TAG 62
#define STOP_TAG  63
#define NEGV  (-10000.0f)

#if __has_builtin(__builtin_amdgcn_global_load_async_to_lds_b128) && \
    __has_builtin(__builtin_amdgcn_s_wait_asynccnt)
#define HAVE_ASYNC_LDS 1
typedef int v4i __attribute__((vector_size(16)));
typedef __attribute__((address_space(1))) v4i gv4i_t;   // global (AS1)
typedef __attribute__((address_space(3))) v4i lv4i_t;   // LDS (AS3)
#else
#define HAVE_ASYNC_LDS 0
#endif

// ---------------------------------------------------------------------------
// Fragment helpers (CDNA5 wave32 WMMA 16x16x32 bf16 layouts, ISA 7.12.2)
// A 16-bit 16x32: lane L holds row (L&15); halves 0..7 = K kb..kb+7,
//                 halves 8..15 = K 16+kb..16+kb+7, kb = 8*(L>=16)
// B: column n = row n of row-major [N,K] weight; same K order as A (kb, 16+kb)
// ---------------------------------------------------------------------------
__device__ __forceinline__ v16bf load_a_frag(const bf16_t* rowptr, int k0, int kb) {
  union { v16bf v; v8bf h[2]; } u;
  u.h[0] = *(const v8bf*)(rowptr + k0 + kb);
  u.h[1] = *(const v8bf*)(rowptr + k0 + kb + 16);
  return u.v;
}
__device__ __forceinline__ v16bf load_b_frag(const bf16_t* rowptr, int k0, int ko) {
  union { v16bf v; v8bf h[2]; } u;
  u.h[0] = *(const v8bf*)(rowptr + k0 + (ko >> 1));
  u.h[1] = *(const v8bf*)(rowptr + k0 + (ko >> 1) + 16);
  return u.v;
}
__device__ __forceinline__ v8f wmma_bf16(v16bf a, v16bf b, v8f c) {
  return __builtin_amdgcn_wmma_f32_16x16x32_bf16(false, a, false, b, (short)0, c, false, false);
}

// ---------------------------------------------------------------------------
// K0: f32 -> bf16 conversion
// ---------------------------------------------------------------------------
__global__ void cvt_f32_bf16(const float* __restrict__ src, bf16_t* __restrict__ dst, int n) {
  int i = blockIdx.x * blockDim.x + threadIdx.x;
  if (i < n) dst[i] = (bf16_t)src[i];
}

// ---------------------------------------------------------------------------
// K1: xg[dir] = X(rev for dir=1) @ W_ih^T + b_ih   (M=S, N=G3, K=E)
// Register-tiled: 64x32 output per wave (4 M-frags x 2 N-frags, 8 accums).
// Per K-step: 4 A frags + 2 B frags (6 KB) feed 8 WMMAs -> 0.75 KB/WMMA.
// ---------------------------------------------------------------------------
__global__ __launch_bounds__(256) void gemm_xg(
    const bf16_t* __restrict__ Xb,   // [S,E] bf16
    const bf16_t* __restrict__ Wf,   // [G3,E] bf16
    const bf16_t* __restrict__ Wb,
    const float*  __restrict__ bihf,
    const float*  __restrict__ bihb,
    float* __restrict__ xgf,         // [S,G3] scan order
    float* __restrict__ xgb)         // [S,G3] scan order (row t = x[S-1-t])
{
  const int MT = S_LEN / 64;         // 256
  const int NT = G3 / 32;            // 48
  int wid  = blockIdx.x * (blockDim.x >> 5) + (threadIdx.x >> 5);
  int lane = threadIdx.x & 31;
  int total = MT * NT;
  int dir = wid / total;
  if (dir > 1) return;
  int rem = wid % total;
  int mt = rem / NT, nt = rem % NT;

  const bf16_t* W    = dir ? Wb   : Wf;
  const float*  bias = dir ? bihb : bihf;
  float*        out  = dir ? xgb  : xgf;

  int r  = lane & 15;
  int kb = (lane >> 4) << 3;
  int ko = (lane >> 4) << 4;

  const bf16_t* aptr[4];
#pragma unroll
  for (int j = 0; j < 4; ++j) {
    int arow = mt * 64 + j * 16 + r;
    if (dir) arow = S_LEN - 1 - arow;
    aptr[j] = Xb + (size_t)arow * E_DIM;
  }
  const bf16_t* bptr0 = W + (size_t)(nt * 32 + r)      * E_DIM;
  const bf16_t* bptr1 = W + (size_t)(nt * 32 + 16 + r) * E_DIM;

  v8f acc[4][2] = {};
#pragma unroll 4
  for (int k0 = 0; k0 < E_DIM; k0 += 32) {
    v16bf b0 = load_b_frag(bptr0, k0, ko);
    v16bf b1 = load_b_frag(bptr1, k0, ko);
#pragma unroll
    for (int j = 0; j < 4; ++j) {
      v16bf a = load_a_frag(aptr[j], k0, kb);
      acc[j][0] = wmma_bf16(a, b0, acc[j][0]);
      acc[j][1] = wmma_bf16(a, b1, acc[j][1]);
    }
  }

  int nn    = lane & 15;
  int mhalf = (lane >> 4) ? 8 : 0;
#pragma unroll
  for (int j = 0; j < 4; ++j) {
    int mbase = mt * 64 + j * 16 + mhalf;
#pragma unroll
    for (int q = 0; q < 2; ++q) {
      int n = nt * 32 + q * 16 + nn;
      float bv = bias[n];
#pragma unroll
      for (int i = 0; i < 8; ++i)
        out[(size_t)(mbase + i) * G3 + n] = acc[j][q][i] + bv;
    }
  }
}

// ---------------------------------------------------------------------------
// K2: sequential GRU scan, one workgroup per direction (persistent).
// hg = h @ W_hh^T via WMMA with h broadcast into every A row; h lives in LDS.
// Next timestep's xg row (6 KB) is double-buffered into LDS with
// GLOBAL_LOAD_ASYNC_TO_LDS_B128 (ASYNCcnt) during the WMMA phase.
// ---------------------------------------------------------------------------
__global__ __launch_bounds__(1024) void gru_scan(
    const float*  __restrict__ h0,    // [2,HD]
    const bf16_t* __restrict__ Whhf,  // [G3,HD] bf16
    const bf16_t* __restrict__ Whhb,
    const float*  __restrict__ bhhf,
    const float*  __restrict__ bhhb,
    const float*  __restrict__ xgf,   // [S,G3]
    const float*  __restrict__ xgb,
    bf16_t* __restrict__ outc)        // [S,H] bf16, cols 0..511 fwd, 512..1023 bwd
{
  __shared__ float  h_f32[HD];
  __shared__ bf16_t h_bf[HD];
  __shared__ float  hg[G3];
  __shared__ float  bhh_s[G3];
#if HAVE_ASYNC_LDS
  __shared__ float  xg_s[2][G3];      // double buffer, 12 KB
#endif

  int dir = blockIdx.x;
  const bf16_t* Whh = dir ? Whhb : Whhf;
  const float*  bhh = dir ? bhhb : bhhf;
  const float*  xg  = dir ? xgb  : xgf;

  int tid  = threadIdx.x;
  int lane = tid & 31;
  int wave = tid >> 5;               // 0..31  (96 N-tiles -> 3 per wave)

  if (tid < HD) { float v = h0[dir * HD + tid]; h_f32[tid] = v; h_bf[tid] = (bf16_t)v; }
  for (int i = tid; i < G3; i += blockDim.x) bhh_s[i] = bhh[i];

#if HAVE_ASYNC_LDS
  // Stage xg row 0 into buffer 0 (384 lanes x 16B = 6 KB).
  if (tid < 384) {
    __builtin_amdgcn_global_load_async_to_lds_b128(
        (gv4i_t*)(xg + (size_t)tid * 4),
        (lv4i_t*)(&xg_s[0][tid * 4]), 0, 0);
    __builtin_amdgcn_s_wait_asynccnt(0);
  }
#endif
  __syncthreads();

  int r  = lane & 15;
  int kb = (lane >> 4) << 3;
  int ko = (lane >> 4) << 4;

  for (int t = 0; t < S_LEN; ++t) {
#if HAVE_ASYNC_LDS
    // Kick off the async stage of xg[t+1] into the other buffer.
    if (t + 1 < S_LEN && tid < 384) {
      __builtin_amdgcn_global_load_async_to_lds_b128(
          (gv4i_t*)(xg + (size_t)(t + 1) * G3 + tid * 4),
          (lv4i_t*)(&xg_s[(t + 1) & 1][tid * 4]), 0, 0);
    }
#else
    if (t + 1 < S_LEN && tid < 48)
      __builtin_prefetch(xg + (size_t)(t + 1) * G3 + tid * 32, 0, 1);
#endif

#pragma unroll
    for (int tt = 0; tt < 3; ++tt) {
      int nt = wave * 3 + tt;
      const bf16_t* bptr = Whh + (size_t)(nt * 16 + r) * HD;
      v8f acc = {};
#pragma unroll
      for (int k0 = 0; k0 < HD; k0 += 32) {
        v16bf a = load_a_frag(h_bf, k0, kb);   // every lane loads h -> all rows = h
        v16bf b = load_b_frag(bptr, k0, ko);
        acc = wmma_bf16(a, b, acc);
      }
      if (lane < 16) hg[nt * 16 + lane] = acc[0];   // D row 0 (all rows identical)
    }

#if HAVE_ASYNC_LDS
    // Async loads complete in order: <=1 outstanding means the buffer for
    // step t has landed while step t+1's stage may still be in flight.
    if (t + 1 < S_LEN) {
      if (tid < 384) __builtin_amdgcn_s_wait_asynccnt(1);
    } else {
      if (tid < 384) __builtin_amdgcn_s_wait_asynccnt(0);
    }
#endif
    __syncthreads();

    if (tid < HD) {
#if HAVE_ASYNC_LDS
      const float* xr = &xg_s[t & 1][0];
#else
      const float* xr = xg + (size_t)t * G3;
#endif
      float ir  = xr[tid];
      float iz  = xr[HD + tid];
      float inn = xr[2 * HD + tid];
      float hr = hg[tid]          + bhh_s[tid];
      float hz = hg[HD + tid]     + bhh_s[HD + tid];
      float hn = hg[2 * HD + tid] + bhh_s[2 * HD + tid];
      float rg = 1.0f / (1.0f + __expf(-(ir + hr)));
      float zg = 1.0f / (1.0f + __expf(-(iz + hz)));
      float ng = tanhf(inn + rg * hn);
      float hnew = (1.0f - zg) * ng + zg * h_f32[tid];
      h_f32[tid] = hnew;
      h_bf[tid]  = (bf16_t)hnew;
      int row = dir ? (S_LEN - 1 - t) : t;          // backward written back in time order
      outc[(size_t)row * H_DIM + dir * HD + tid] = (bf16_t)hnew;
    }
    __syncthreads();
  }
}

// ---------------------------------------------------------------------------
// K3: feat = outc @ proj_w^T + proj_b   (M=S, N=64, K=1024)
// One 16x64 tile per wave: A fragment reused across all 4 N-tiles.
// ---------------------------------------------------------------------------
__global__ __launch_bounds__(256) void gemm_proj(
    const bf16_t* __restrict__ Ab,   // [S,H] bf16
    const bf16_t* __restrict__ Pw,   // [NTAG,H] bf16
    const float*  __restrict__ pb,
    float* __restrict__ feat)        // [S,NTAG]
{
  int wid  = blockIdx.x * (blockDim.x >> 5) + (threadIdx.x >> 5);
  int lane = threadIdx.x & 31;
  int mt = wid;
  if (mt >= S_LEN / 16) return;

  int r  = lane & 15;
  int kb = (lane >> 4) << 3;
  int ko = (lane >> 4) << 4;
  const bf16_t* aptr = Ab + (size_t)(mt * 16 + r) * H_DIM;
  const bf16_t* bptr[4];
#pragma unroll
  for (int q = 0; q < 4; ++q)
    bptr[q] = Pw + (size_t)(q * 16 + r) * H_DIM;

  v8f acc[4] = {};
#pragma unroll 4
  for (int k0 = 0; k0 < H_DIM; k0 += 32) {
    v16bf a = load_a_frag(aptr, k0, kb);
#pragma unroll
    for (int q = 0; q < 4; ++q) {
      v16bf b = load_b_frag(bptr[q], k0, ko);
      acc[q] = wmma_bf16(a, b, acc[q]);
    }
  }

  int nn    = lane & 15;
  int mbase = mt * 16 + ((lane >> 4) ? 8 : 0);
#pragma unroll
  for (int q = 0; q < 4; ++q) {
    int n = q * 16 + nn;
    float bv = pb[n];
#pragma unroll
    for (int i = 0; i < 8; ++i)
      feat[(size_t)(mbase + i) * NTAG + n] = acc[q][i] + bv;
  }
}

// ---------------------------------------------------------------------------
// K4: Viterbi forward pass. 64 threads, transitions cached in LDS.
// ---------------------------------------------------------------------------
__global__ __launch_bounds__(64) void viterbi_fwd(
    const float* __restrict__ feat,        // [S,NTAG]
    const float* __restrict__ trans,       // [NTAG,NTAG]
    unsigned char* __restrict__ bp,        // [S,NTAG]
    float* __restrict__ vfinal)            // [NTAG]
{
  __shared__ float Ts[NTAG * NTAG];
  __shared__ float v[NTAG];
  __shared__ float vn[NTAG];
  int tid = threadIdx.x;
  for (int i = tid; i < NTAG * NTAG; i += blockDim.x) Ts[i] = trans[i];
  v[tid] = (tid == START_TAG) ? 0.0f : NEGV;
  __syncthreads();

  for (int t = 0; t < S_LEN; ++t) {
    const float* Ti = Ts + tid * NTAG;
    float best = -3.4e38f;
    int bj = 0;
#pragma unroll 8
    for (int j = 0; j < NTAG; ++j) {
      float s = v[j] + Ti[j];
      if (s > best) { best = s; bj = j; }   // strict > keeps first max (argmax semantics)
    }
    bp[(size_t)t * NTAG + tid] = (unsigned char)bj;
    vn[tid] = best + feat[(size_t)t * NTAG + tid];
    __syncthreads();
    v[tid] = vn[tid];
    __syncthreads();
  }
  vfinal[tid] = v[tid];
}

// ---------------------------------------------------------------------------
// K5: terminal argmax + backtrack (sequential pointer chase in L2)
// out[0] = path_score, out[1..S] = path (as float)
// ---------------------------------------------------------------------------
__global__ void viterbi_back(
    const float* __restrict__ vfinal,
    const float* __restrict__ trans,
    const unsigned char* __restrict__ bp,
    float* __restrict__ out)
{
  if (threadIdx.x != 0 || blockIdx.x != 0) return;
  float best = -3.4e38f;
  int bl = 0;
  for (int j = 0; j < NTAG; ++j) {
    float s = vfinal[j] + trans[STOP_TAG * NTAG + j];
    if (s > best) { best = s; bl = j; }
  }
  out[0] = best;
  int state = bl;
  out[1 + (S_LEN - 1)] = (float)state;
  for (int t = S_LEN - 2; t >= 0; --t) {
    state = bp[(size_t)(t + 1) * NTAG + state];
    out[1 + t] = (float)state;
  }
}

// ---------------------------------------------------------------------------
extern "C" void kernel_launch(void* const* d_in, const int* in_sizes, int n_in,
                              void* d_out, int out_size, void* d_ws, size_t ws_size,
                              hipStream_t stream) {
  (void)in_sizes; (void)n_in; (void)out_size; (void)ws_size;

  const float* sentence = (const float*)d_in[0];   // [S,1,E]
  const float* h0       = (const float*)d_in[1];   // [2,1,HD]
  const float* w_ih_f   = (const float*)d_in[2];   // [G3,E]
  const float* w_hh_f   = (const float*)d_in[3];   // [G3,HD]
  const float* b_ih_f   = (const float*)d_in[4];
  const float* b_hh_f   = (const float*)d_in[5];
  const float* w_ih_b   = (const float*)d_in[6];
  const float* w_hh_b   = (const float*)d_in[7];
  const float* b_ih_b   = (const float*)d_in[8];
  const float* b_hh_b   = (const float*)d_in[9];
  const float* proj_w   = (const float*)d_in[10];  // [NTAG,H]
  const float* proj_b   = (const float*)d_in[11];
  const float* trans    = (const float*)d_in[12];  // [NTAG,NTAG]
  float* out = (float*)d_out;

  char* ws = (char*)d_ws;
  size_t off = 0;
  auto alloc = [&](size_t bytes) -> void* {
    void* p = ws + off;
    off = (off + bytes + 255) & ~(size_t)255;
    return p;
  };
  bf16_t* Xb    = (bf16_t*)alloc((size_t)S_LEN * E_DIM * sizeof(bf16_t));
  bf16_t* Wihf  = (bf16_t*)alloc((size_t)G3 * E_DIM * sizeof(bf16_t));
  bf16_t* Wihb  = (bf16_t*)alloc((size_t)G3 * E_DIM * sizeof(bf16_t));
  bf16_t* Whhf  = (bf16_t*)alloc((size_t)G3 * HD * sizeof(bf16_t));
  bf16_t* Whhb  = (bf16_t*)alloc((size_t)G3 * HD * sizeof(bf16_t));
  bf16_t* PwB   = (bf16_t*)alloc((size_t)NTAG * H_DIM * sizeof(bf16_t));
  float*  xgf   = (float*)alloc((size_t)S_LEN * G3 * sizeof(float));
  float*  xgb   = (float*)alloc((size_t)S_LEN * G3 * sizeof(float));
  bf16_t* outc  = (bf16_t*)alloc((size_t)S_LEN * H_DIM * sizeof(bf16_t));
  float*  feat  = (float*)alloc((size_t)S_LEN * NTAG * sizeof(float));
  unsigned char* bp = (unsigned char*)alloc((size_t)S_LEN * NTAG);
  float*  vfin  = (float*)alloc(NTAG * sizeof(float));

  auto cvt = [&](const float* s, bf16_t* d, int n) {
    cvt_f32_bf16<<<(n + 255) / 256, 256, 0, stream>>>(s, d, n);
  };
  cvt(sentence, Xb,   S_LEN * E_DIM);
  cvt(w_ih_f,   Wihf, G3 * E_DIM);
  cvt(w_ih_b,   Wihb, G3 * E_DIM);
  cvt(w_hh_f,   Whhf, G3 * HD);
  cvt(w_hh_b,   Whhb, G3 * HD);
  cvt(proj_w,   PwB,  NTAG * H_DIM);

  // K1: 2 dirs * (S/64)*(G3/32) = 24576 wave-tiles, 8 waves/block
  {
    int waves = 2 * (S_LEN / 64) * (G3 / 32);
    gemm_xg<<<waves / 8, 256, 0, stream>>>(Xb, Wihf, Wihb, b_ih_f, b_ih_b, xgf, xgb);
  }
  // K2: persistent scan, one workgroup per direction
  gru_scan<<<2, 1024, 0, stream>>>(h0, Whhf, Whhb, b_hh_f, b_hh_b, xgf, xgb, outc);
  // K3: projection, (S/16) wave-tiles of 16x64
  {
    int waves = S_LEN / 16;
    gemm_proj<<<waves / 8, 256, 0, stream>>>(outc, PwB, proj_b, feat);
  }
  // K4/K5: Viterbi
  viterbi_fwd<<<1, 64, 0, stream>>>(feat, trans, bp, vfin);
  viterbi_back<<<1, 32, 0, stream>>>(vfin, trans, bp, out);
}
